// ClusterTransformerBlock_40999757807820
// MI455X (gfx1250) — compile-verified, matrix operand order checked by reading, not run on hardware
//
#include <hip/hip_runtime.h>
#include <hip/hip_bf16.h>

#define NTOK 16384
#define NB   8192
#define CDIM 256
#define HH   8
#define CHD  32
#define MM   32
#define HIDD 512
#define TABN 3025
#define POSF 5

typedef __attribute__((ext_vector_type(16))) __bf16 v16bf;
typedef __attribute__((ext_vector_type(8)))  float  v8f;

union FragBF { uint4 u4[2]; v16bf v; };

__device__ __forceinline__ float wave_max32(float v) {
  #pragma unroll
  for (int o = 16; o > 0; o >>= 1) v = fmaxf(v, __shfl_xor(v, o, 32));
  return v;
}
__device__ __forceinline__ float wave_sum32(float v) {
  #pragma unroll
  for (int o = 16; o > 0; o >>= 1) v += __shfl_xor(v, o, 32);
  return v;
}

// ---------------- weight fp32 -> bf16 ----------------
__global__ void cvt_bf16_kernel(const float* __restrict__ src,
                                __hip_bfloat16* __restrict__ dst, int n) {
  int i = blockIdx.x * blockDim.x + threadIdx.x;
  if (i < n) dst[i] = __float2bfloat16(src[i]);
}

// ---------------- positional table: pre_table @ pe_w.T + pe_b ----------------
__global__ void pe_table_kernel(const float* __restrict__ pre,
                                const float* __restrict__ pew,
                                const float* __restrict__ peb,
                                float* __restrict__ tab) {
  int i = blockIdx.x * blockDim.x + threadIdx.x;
  if (i >= TABN * HH) return;
  int t = i / HH, h = i % HH;
  float acc = peb[h];
  #pragma unroll
  for (int f = 0; f < POSF; ++f) acc += pre[t * POSF + f] * pew[h * POSF + f];
  tab[i] = acc;
}

// ---------------- LayerNorm (wave per token) -> bf16 ----------------
__global__ void ln_bf16_kernel(const float* __restrict__ x,
                               const float* __restrict__ w,
                               const float* __restrict__ b,
                               __hip_bfloat16* __restrict__ out) {
  int token = blockIdx.x * 8 + (threadIdx.x >> 5);
  int lane  = threadIdx.x & 31;
  const float* xr = x + (size_t)token * CDIM + lane * 8;
  float v[8];
  *(float4*)&v[0] = *(const float4*)&xr[0];
  *(float4*)&v[4] = *(const float4*)&xr[4];
  float s = 0.f, q = 0.f;
  #pragma unroll
  for (int j = 0; j < 8; ++j) { s += v[j]; q += v[j] * v[j]; }
  s = wave_sum32(s); q = wave_sum32(q);
  float mu  = s * (1.0f / CDIM);
  float var = q * (1.0f / CDIM) - mu * mu;
  float rstd = rsqrtf(var + 1e-5f);
  __hip_bfloat16* o = out + (size_t)token * CDIM + lane * 8;
  #pragma unroll
  for (int j = 0; j < 8; ++j) {
    int c = lane * 8 + j;
    o[j] = __float2bfloat16((v[j] - mu) * rstd * w[c] + b[c]);
  }
}

// ---------------- bf16 WMMA GEMM:  out[n,o] = sum_k A[n,k]*W[o,k] (+epilogue) ----
// Block = 8 waves = 128 rows x 64 cols. The 64-col W strip is staged in LDS
// (shared by all 8 waves, padded rows -> conflict-free ds_load_b128 fragments);
// A fragments are register double-buffered from global (2 VMEM loads/K-step).
enum { EPI_Q = 0, EPI_PLAIN = 1, EPI_RESID = 2, EPI_GELU = 3 };

template <int EPI, int K>
__global__ void gemm_bf16_kernel(const __hip_bfloat16* __restrict__ A,
                                 const __hip_bfloat16* __restrict__ W,
                                 const float* __restrict__ bias,
                                 const float* __restrict__ resid,
                                 float* __restrict__ outF,
                                 __hip_bfloat16* __restrict__ outH,
                                 int OutF, float scale) {
  constexpr int KBLK = 256;
  constexpr int SWK  = KBLK + 8;            // +16B row pad: banks rotate by 4/row
  __shared__ __hip_bfloat16 sW[64 * SWK];   // 33.8 KB

  int wv   = threadIdx.x >> 5;
  int lane = threadIdx.x & 31;
  int rowBase = blockIdx.x * 128 + wv * 16;
  int colBase = blockIdx.y * 64;
  int koff = (lane >> 4) << 3;              // lanes 16-31 take K+8 / K+24 halves
  const __hip_bfloat16* arow = A + (size_t)(rowBase + (lane & 15)) * K + koff;

  v8f acc[4] = {v8f{}, v8f{}, v8f{}, v8f{}};
  FragBF fa[2];

  #pragma unroll
  for (int kb = 0; kb < K; kb += KBLK) {
    // ---- cooperative stage of W[colBase..colBase+63][kb..kb+255] into LDS ----
    #pragma unroll
    for (int c = threadIdx.x; c < 64 * (KBLK / 8); c += 256) {
      int row = c >> 5;                     // KBLK/8 == 32 chunks per row
      int off = (c & 31) << 3;
      *(uint4*)&sW[row * SWK + off] =
          *(const uint4*)(W + (size_t)(colBase + row) * K + kb + off);
    }
    __syncthreads();

    // preload A fragment for first K-step of this block
    fa[0].u4[0] = *(const uint4*)(arow + kb);
    fa[0].u4[1] = *(const uint4*)(arow + kb + 16);

    #pragma unroll
    for (int k0 = 0; k0 < KBLK; k0 += 32) {
      const int cur = (k0 >> 5) & 1;
      const int nxt = cur ^ 1;
      if (k0 + 32 < KBLK) {                 // double-buffer next A fragment
        fa[nxt].u4[0] = *(const uint4*)(arow + kb + k0 + 32);
        fa[nxt].u4[1] = *(const uint4*)(arow + kb + k0 + 48);
      }
      #pragma unroll
      for (int t = 0; t < 4; ++t) {
        const __hip_bfloat16* wl = sW + ((lane & 15) + 16 * t) * SWK + koff + k0;
        FragBF fbt;
        fbt.u4[0] = *(const uint4*)(wl);
        fbt.u4[1] = *(const uint4*)(wl + 16);
        acc[t] = __builtin_amdgcn_wmma_f32_16x16x32_bf16(
            /*neg_a=*/false, fa[cur].v, /*neg_b=*/false, fbt.v,
            /*c_mod=*/(short)0, acc[t], /*reuse_a=*/false, /*reuse_b=*/false);
      }
    }
    __syncthreads();
  }

  int lh = (lane >> 4) * 8;
  int ln = lane & 15;
  #pragma unroll
  for (int t = 0; t < 4; ++t) {
    int col = colBase + t * 16 + ln;
    float bv = bias[col];
    #pragma unroll
    for (int r = 0; r < 8; ++r) {
      int row = rowBase + r + lh;
      float val = acc[t][r] + bv;
      size_t o = (size_t)row * OutF + col;
      if (EPI == EPI_Q) {
        outF[o] = val * scale;
      } else if (EPI == EPI_PLAIN) {
        outF[o] = val;
      } else if (EPI == EPI_RESID) {
        outF[o] = val + resid[o];
      } else {  // GELU (exact) -> bf16
        float g = 0.5f * val * (1.0f + erff(val * 0.70710678118f));
        outH[o] = __float2bfloat16(g);
      }
    }
  }
}

// ---------------- gathered cluster attention (wave per query) ----------------
__global__ void attn_kernel(const float* __restrict__ q,
                            const float* __restrict__ kv,
                            const int* __restrict__ midx,
                            const float* __restrict__ cmask,
                            const int* __restrict__ peidx,
                            const float* __restrict__ petab,
                            const float* __restrict__ blank_k,
                            const float* __restrict__ blank_v,
                            __hip_bfloat16* __restrict__ out) {
  __shared__ float s_q[8][CDIM];
  __shared__ float s_p[8][HH][MM];
  __shared__ float s_pb[8][HH];
  __shared__ float s_bl[8][HH];
  __shared__ int   s_idx[8][MM];

  int wv = threadIdx.x >> 5, lane = threadIdx.x & 31;
  int t = blockIdx.x * 8 + wv;
  size_t tb = (size_t)t;

  const float* qr = q + tb * CDIM + lane * 8;
  *(float4*)&s_q[wv][lane * 8]     = *(const float4*)&qr[0];
  *(float4*)&s_q[wv][lane * 8 + 4] = *(const float4*)&qr[4];
  int idxv = midx[tb * MM + lane];
  s_idx[wv][lane] = idxv;
  int pv = peidx[tb * MM + lane];
  float madd = (1.0f - cmask[tb * MM + lane]) * -100.0f;
  __syncthreads();

  // blank logits, one head per lane 0..7
  if (lane < HH) {
    float acc = 0.f;
    #pragma unroll
    for (int c = 0; c < CHD; ++c) acc += s_q[wv][lane * CHD + c] * blank_k[lane * CHD + c];
    s_bl[wv][lane] = fminf(5.0f, fmaxf(-5.0f, acc));
  }

  // member logits, lane = member m
  size_t bbase = (size_t)(t & ~(NB - 1));          // b*N  (N power of two)
  size_t krow  = (bbase + (size_t)idxv) * (2 * CDIM);
  const float* pt = petab + (size_t)pv * HH;
  float logit[HH];
  #pragma unroll
  for (int h = 0; h < HH; ++h) {
    const float* kp = kv + krow + h * (2 * CHD);
    float acc = 0.f;
    #pragma unroll
    for (int c = 0; c < CHD; c += 4) {
      float4 kk = *(const float4*)&kp[c];
      acc += s_q[wv][h * CHD + c + 0] * kk.x + s_q[wv][h * CHD + c + 1] * kk.y +
             s_q[wv][h * CHD + c + 2] * kk.z + s_q[wv][h * CHD + c + 3] * kk.w;
    }
    logit[h] = acc + pt[h] + madd;
  }
  __syncthreads();

  // softmax over 32 members + blank (wave32 shuffles)
  #pragma unroll
  for (int h = 0; h < HH; ++h) {
    float bl = s_bl[wv][h];
    float mx = fmaxf(wave_max32(logit[h]), bl);
    float e  = __expf(logit[h] - mx);
    float es = wave_sum32(e);
    float eb = __expf(bl - mx);
    float inv = 1.0f / (es + eb);
    s_p[wv][h][lane] = e * inv;
    if (lane == 0) s_pb[wv][h] = eb * inv;
  }
  __syncthreads();

  // output accumulation, lane = channel within head (coalesced V gather)
  __hip_bfloat16* orow = out + tb * CDIM;
  #pragma unroll
  for (int h = 0; h < HH; ++h) {
    float acc = s_pb[wv][h] * blank_v[h * CHD + lane];
    for (int m = 0; m < MM; ++m) {
      size_t vrow = (bbase + (size_t)s_idx[wv][m]) * (2 * CDIM);
      acc += s_p[wv][h][m] * kv[vrow + h * (2 * CHD) + CHD + lane];
    }
    orow[h * CHD + lane] = __float2bfloat16(acc);
  }
}

extern "C" void kernel_launch(void* const* d_in, const int* in_sizes, int n_in,
                              void* d_out, int out_size, void* d_ws, size_t ws_size,
                              hipStream_t stream) {
  (void)in_sizes; (void)n_in; (void)out_size; (void)ws_size;
  const float* feat      = (const float*)d_in[0];
  const int*   midx      = (const int*)d_in[1];
  const float* cmask     = (const float*)d_in[2];
  const int*   peidx     = (const int*)d_in[3];
  const float* pre_table = (const float*)d_in[5];
  const float* n1w  = (const float*)d_in[6];
  const float* n1b  = (const float*)d_in[7];
  const float* qw   = (const float*)d_in[8];
  const float* qb   = (const float*)d_in[9];
  const float* kvw  = (const float*)d_in[10];
  const float* kvb  = (const float*)d_in[11];
  const float* blank_k = (const float*)d_in[12];
  const float* blank_v = (const float*)d_in[13];
  const float* pew  = (const float*)d_in[14];
  const float* peb  = (const float*)d_in[15];
  const float* pjw  = (const float*)d_in[16];
  const float* pjb  = (const float*)d_in[17];
  const float* n2w  = (const float*)d_in[18];
  const float* n2b  = (const float*)d_in[19];
  const float* f1w  = (const float*)d_in[20];
  const float* f1b  = (const float*)d_in[21];
  const float* f2w  = (const float*)d_in[22];
  const float* f2b  = (const float*)d_in[23];
  float* outp = (float*)d_out;

  char* ws = (char*)d_ws;
  size_t off = 0;
  auto alloc = [&](size_t bytes) -> void* {
    void* p = ws + off;
    off = (off + bytes + 255) & ~(size_t)255;
    return p;
  };
  __hip_bfloat16* wq  = (__hip_bfloat16*)alloc((size_t)CDIM * CDIM * 2);
  __hip_bfloat16* wkv = (__hip_bfloat16*)alloc((size_t)2 * CDIM * CDIM * 2);
  __hip_bfloat16* wpj = (__hip_bfloat16*)alloc((size_t)CDIM * CDIM * 2);
  __hip_bfloat16* wf1 = (__hip_bfloat16*)alloc((size_t)HIDD * CDIM * 2);
  __hip_bfloat16* wf2 = (__hip_bfloat16*)alloc((size_t)CDIM * HIDD * 2);
  float*          petab = (float*)alloc((size_t)TABN * HH * 4);
  __hip_bfloat16* xln   = (__hip_bfloat16*)alloc((size_t)NTOK * CDIM * 2); // also attn-out
  float*          qbuf  = (float*)alloc((size_t)NTOK * CDIM * 4);          // also x-residual
  float*          kvbuf = (float*)alloc((size_t)NTOK * 2 * CDIM * 4);      // also yln + hidden
  __hip_bfloat16* aout = xln;
  float*          xres = qbuf;
  __hip_bfloat16* yln  = (__hip_bfloat16*)kvbuf;
  __hip_bfloat16* hid  = (__hip_bfloat16*)((char*)kvbuf + (size_t)NTOK * CDIM * 2);

  const int thr = 256;
  // weight conversion + positional table
  cvt_bf16_kernel<<<(CDIM * CDIM) / 256, thr, 0, stream>>>(qw, wq, CDIM * CDIM);
  cvt_bf16_kernel<<<(2 * CDIM * CDIM) / 256, thr, 0, stream>>>(kvw, wkv, 2 * CDIM * CDIM);
  cvt_bf16_kernel<<<(CDIM * CDIM) / 256, thr, 0, stream>>>(pjw, wpj, CDIM * CDIM);
  cvt_bf16_kernel<<<(HIDD * CDIM) / 256, thr, 0, stream>>>(f1w, wf1, HIDD * CDIM);
  cvt_bf16_kernel<<<(CDIM * HIDD) / 256, thr, 0, stream>>>(f2w, wf2, CDIM * HIDD);
  pe_table_kernel<<<(TABN * HH + 255) / 256, thr, 0, stream>>>(pre_table, pew, peb, petab);

  // LN1 -> bf16
  ln_bf16_kernel<<<NTOK / 8, thr, 0, stream>>>(feat, n1w, n1b, xln);

  // Q and KV projections (WMMA bf16)
  dim3 gq(NTOK / 128, CDIM / 64);
  gemm_bf16_kernel<EPI_Q, CDIM><<<gq, thr, 0, stream>>>(xln, wq, qb, nullptr, qbuf, nullptr,
                                                        CDIM, 0.17677669529663689f);
  dim3 gkv(NTOK / 128, (2 * CDIM) / 64);
  gemm_bf16_kernel<EPI_PLAIN, CDIM><<<gkv, thr, 0, stream>>>(xln, wkv, kvb, nullptr, kvbuf,
                                                             nullptr, 2 * CDIM, 1.0f);

  // gathered attention (writes bf16 into aout, aliasing xln)
  attn_kernel<<<NTOK / 8, thr, 0, stream>>>(qbuf, kvbuf, midx, cmask, peidx, petab,
                                            blank_k, blank_v, aout);

  // output projection + residual (x = feat + proj)
  gemm_bf16_kernel<EPI_RESID, CDIM><<<gq, thr, 0, stream>>>(aout, wpj, pjb, feat, xres,
                                                            nullptr, CDIM, 1.0f);

  // LN2 -> bf16
  ln_bf16_kernel<<<NTOK / 8, thr, 0, stream>>>(xres, n2w, n2b, yln);

  // MLP: fc1 + exact GELU -> bf16, fc2 + residual -> final output
  dim3 gf1(NTOK / 128, HIDD / 64);
  gemm_bf16_kernel<EPI_GELU, CDIM><<<gf1, thr, 0, stream>>>(yln, wf1, f1b, nullptr, nullptr,
                                                            hid, HIDD, 1.0f);
  dim3 gf2(NTOK / 128, CDIM / 64);
  gemm_bf16_kernel<EPI_RESID, HIDD><<<gf2, thr, 0, stream>>>(hid, wf2, f2b, xres, outp,
                                                             nullptr, CDIM, 1.0f);
}
